// YOLOXLoss_3813930959029
// MI455X (gfx1250) — compile-verified, matrix operand order checked by reading, not run on hardware
//
#include <hip/hip_runtime.h>
#include <hip/hip_bf16.h>
#include <stdint.h>

typedef unsigned long long u64;

#define NUMB 16
#define NUMG 50
#define NUMA 8400
#define CRAD 2.5f
#define SL1_BETA (1.0f/9.0f)

typedef float    v8f   __attribute__((ext_vector_type(8)));
typedef float    v2f   __attribute__((ext_vector_type(2)));
typedef _Float16 v16h  __attribute__((ext_vector_type(16)));
typedef unsigned int u32x4 __attribute__((ext_vector_type(4)));
typedef int      i32x4 __attribute__((ext_vector_type(4)));
typedef int      i32x8 __attribute__((ext_vector_type(8)));

#if __has_builtin(__builtin_amdgcn_tensor_load_to_lds) && __has_builtin(__builtin_amdgcn_s_wait_tensorcnt)
#define HAVE_TDM 1
#else
#define HAVE_TDM 0
#endif

__device__ __forceinline__ float softplusf(float x){
  return fmaxf(x, 0.f) + log1pf(expf(-fabsf(x)));
}
__device__ __forceinline__ float sl1(float d){
  float a = fabsf(d);
  return (a < SL1_BETA) ? 0.5f * a * a / SL1_BETA : a - 0.5f * SL1_BETA;
}
__device__ __forceinline__ unsigned fltord(float f){
  unsigned u = __float_as_uint(f);
  return (u & 0x80000000u) ? ~u : (u | 0x80000000u);
}
__device__ __forceinline__ float ordflt(unsigned o){
  unsigned u = (o & 0x80000000u) ? (o ^ 0x80000000u) : ~o;
  return __uint_as_float(u);
}
// anchor index -> (level, x, y, stride, level width)
__device__ __forceinline__ void adec(int a, int& lvl, int& x, int& y, float& s, int& W){
  if (a < 6400)      { lvl = 0; W = 80; y = a / 80;              x = a - 80 * y;        s = 8.f;  }
  else if (a < 8000) { lvl = 1; W = 40; int i = a - 6400; y = i / 40; x = i - 40 * y;   s = 16.f; }
  else               { lvl = 2; W = 20; int i = a - 8000; y = i / 20; x = i - 20 * y;   s = 32.f; }
}

// ---- Tensor Data Mover: DMA `nelem` f32 from global -> LDS offset 0 (1D tile) ----
// 6-arg builtin form (clang-23 / therock-10.0 headers):
//   (uint32x4 g0, int32x8 g1, int32x4 g2, int32x4 g3, int32x8 g4, i32 cpol)
__device__ __forceinline__ void tdm_load_to_lds0(const float* gptr, int nelem){
#if HAVE_TDM
  unsigned long long ga = (unsigned long long)(uintptr_t)gptr;
  // D# group 0: count=1, lds_addr=0, 57-bit global addr, type=2 ("image")
  u32x4 g0 = { 1u,
               0u,
               (unsigned)(ga & 0xFFFFFFFFull),
               (unsigned)((ga >> 32) & 0x01FFFFFFull) | (2u << 30) };
  // D# group 1: data_size=4B (code 2), tensor_dim0=nelem, tensor_dim1=1,
  //             tile_dim0=nelem, tile_dim1/2=0, tensor_dim0_stride=nelem
  i32x8 g1 = { (int)(2u << 16),
               (int)(((unsigned)nelem & 0xFFFFu) << 16),
               (int)(1u << 16),
               (int)(((unsigned)nelem) << 16),
               0,
               nelem,
               0, 0 };
  i32x4 g2 = {0,0,0,0};
  i32x4 g3 = {0,0,0,0};
  i32x8 g4 = {0,0,0,0,0,0,0,0};
  __builtin_amdgcn_tensor_load_to_lds(g0, g1, g2, g3, g4, 0);
  __builtin_amdgcn_s_wait_tensorcnt(0);
#else
  (void)gptr; (void)nelem;
#endif
}

// ---------------- K1: decode boxes, init winner, zero per-image sums ----------------
__global__ void k_decode(const float* __restrict__ reg0, const float* __restrict__ reg1,
                         const float* __restrict__ reg2,
                         float* __restrict__ boxes, u64* __restrict__ winner,
                         float* __restrict__ sums)
{
  int gid = blockIdx.x * blockDim.x + threadIdx.x;
  if (gid < NUMB * 8) sums[gid] = 0.f;
  if (gid >= NUMB * NUMA) return;
  int b = gid / NUMA, a = gid - b * NUMA;
  int lvl, x, y, W; float s; adec(a, lvl, x, y, s, W);
  const float* rp = (lvl == 0) ? reg0 : (lvl == 1 ? reg1 : reg2);
  size_t plane = (size_t)W * W;
  size_t base  = (size_t)b * 4 * plane + (size_t)y * W + x;
  float r0 = rp[base], r1 = rp[base + plane], r2 = rp[base + 2*plane], r3 = rp[base + 3*plane];
  float cx = (r0 + x) * s, cy = (r1 + y) * s;
  float w  = expf(r2) * s, h = expf(r3) * s;
  float* bx = boxes + (size_t)gid * 4;
  bx[0] = cx - 0.5f*w; bx[1] = cy - 0.5f*h; bx[2] = cx + 0.5f*w; bx[3] = cy + 0.5f*h;
  winner[gid] = ~0ULL;
}

// ---------------- K2: per-anchor fg + in_both(g) bitmask, GTs staged via TDM ----------------
__global__ void k_mask(const float* __restrict__ gt_boxes,
                       const unsigned char* __restrict__ gt_valid,
                       u64* __restrict__ inboth, unsigned* __restrict__ fgm)
{
  __shared__ __align__(16) float s_gt[208];   // 50 GT boxes, at LDS offset 0
  int b = blockIdx.y;
  const float* gp = gt_boxes + (size_t)b * NUMG * 4;
#if HAVE_TDM
  if (threadIdx.x == 0) tdm_load_to_lds0(gp, NUMG * 4);
#else
  for (int i = threadIdx.x; i < NUMG * 4; i += blockDim.x) s_gt[i] = gp[i];
#endif
  __syncthreads();
  int a = blockIdx.x * blockDim.x + threadIdx.x;
  if (a >= NUMA) return;
  int lvl, x, y, W; float s; adec(a, lvl, x, y, s, W);
  float ax = (x + 0.5f) * s, ay = (y + 0.5f) * s, rad = CRAD * s;
  u64 ib = 0; bool anyb = false, anyc = false;
  for (int g = 0; g < NUMG; ++g){
    if (!gt_valid[b * NUMG + g]) continue;
    float x0 = s_gt[4*g+0], y0 = s_gt[4*g+1], x1 = s_gt[4*g+2], y1 = s_gt[4*g+3];
    bool inb = fminf(fminf(ax - x0, x1 - ax), fminf(ay - y0, y1 - ay)) > 0.f;
    float cx = 0.5f * (x0 + x1), cy = 0.5f * (y0 + y1);
    bool inc = fminf(fminf(ax - (cx - rad), (cx + rad) - ax),
                     fminf(ay - (cy - rad), (cy + rad) - ay)) > 0.f;
    anyb |= inb; anyc |= inc;
    if (inb && inc) ib |= (1ULL << g);
  }
  int idx = b * NUMA + a;
  inboth[idx] = ib;
  fgm[idx] = (anyb || anyc) ? 1u : 0u;
}

// ---------------- K3: SimOTA assignment, one block per (gt, image) ----------------
__global__ void k_assign(const float* cls0, const float* cls1, const float* cls2,
                         const float* obj0, const float* obj1, const float* obj2,
                         const float* __restrict__ gt_boxes,
                         const unsigned char* __restrict__ gt_valid,
                         const float* __restrict__ boxes,
                         const unsigned* __restrict__ fgm,
                         const u64* __restrict__ inboth,
                         u64* __restrict__ winner)
{
  int g = blockIdx.x, b = blockIdx.y, tid = threadIdx.x;
  if (!gt_valid[b * NUMG + g]) return;
  __shared__ __align__(16) unsigned char smem[44160];
  float* s_gt    = (float*)smem;              // offset 0: TDM destination
  u64*   s_list  = (u64*)(smem + 1024);       // 256 x 10 cost candidates
  u64*   s_ilist = (u64*)(smem + 21504);      // 256 x 10 iou candidates
  u64*   s_red   = (u64*)(smem + 41984);      // tree-reduce buffer
  u64*   s_round = (u64*)(smem + 44032);      // 10 round winners
#if HAVE_TDM
  if (tid == 0) tdm_load_to_lds0(gt_boxes + (size_t)b * NUMG * 4, NUMG * 4);
#else
  for (int i = tid; i < NUMG * 4; i += blockDim.x) s_gt[i] = gt_boxes[(size_t)b*NUMG*4 + i];
#endif
  for (int i = 0; i < 10; ++i){ s_list[tid*10+i] = ~0ULL; s_ilist[tid*10+i] = ~0ULL; }
  __syncthreads();

  float gx0 = s_gt[4*g], gy0 = s_gt[4*g+1], gx1 = s_gt[4*g+2], gy1 = s_gt[4*g+3];
  float garea = (gx1 - gx0) * (gy1 - gy0);

  for (int a = tid; a < NUMA; a += blockDim.x){
    int idx = b * NUMA + a;
    if (!fgm[idx]) continue;                       // non-fg never match (reference masks them)
    const float* bp = boxes + (size_t)idx * 4;
    float b0 = bp[0], b1 = bp[1], b2 = bp[2], b3 = bp[3];
    float iw = fmaxf(fminf(gx1, b2) - fmaxf(gx0, b0), 0.f);
    float ih = fmaxf(fminf(gy1, b3) - fmaxf(gy0, b1), 0.f);
    float inter = iw * ih;
    float iou = inter / (garea + (b2 - b0) * (b3 - b1) - inter + 1e-8f);
    int lvl, x, y, W; float s; adec(a, lvl, x, y, s, W);
    const float* cp = (lvl == 0) ? cls0 : (lvl == 1 ? cls1 : cls2);
    const float* op = (lvl == 0) ? obj0 : (lvl == 1 ? obj1 : obj2);
    size_t o1 = ((size_t)b * W + y) * W + x;
    float cl = cp[o1], ob = op[o1];
    // t=1 (single class): cls_cost = -clip(log sqrt(sig(c)sig(o)), -100)
    float logp = fmaxf(-0.5f * (softplusf(-cl) + softplusf(-ob)), -100.f);
    bool ibo = (inboth[idx] >> g) & 1ULL;
    float cost = -logp - 3.f * logf(iou + 1e-8f) + (ibo ? 0.f : 1e5f);

    u64 ck = ((u64)fltord(cost) << 32) | ((u64)(unsigned)a << 8) | (unsigned)tid;
    { int wi = 0; u64 wv = s_list[tid*10];
      #pragma unroll
      for (int i = 1; i < 10; ++i){ u64 v = s_list[tid*10+i]; if (v > wv){ wv = v; wi = i; } }
      if (ck < wv) s_list[tid*10+wi] = ck; }
    u64 ik = ((u64)(~fltord(iou)) << 32) | ((u64)(unsigned)a << 8) | (unsigned)tid;
    { int wi = 0; u64 wv = s_ilist[tid*10];
      #pragma unroll
      for (int i = 1; i < 10; ++i){ u64 v = s_ilist[tid*10+i]; if (v > wv){ wv = v; wi = i; } }
      if (ik < wv) s_ilist[tid*10+wi] = ik; }
  }
  __syncthreads();

  // sum of global top-10 ious -> dyn_k
  float isum = 0.f;
  for (int r = 0; r < 10; ++r){
    u64 m = ~0ULL;
    #pragma unroll
    for (int i = 0; i < 10; ++i){ u64 v = s_ilist[tid*10+i]; if (v < m) m = v; }
    s_red[tid] = m; __syncthreads();
    for (int off = 128; off > 0; off >>= 1){
      if (tid < off){ u64 o = s_red[tid + off]; if (o < s_red[tid]) s_red[tid] = o; }
      __syncthreads();
    }
    u64 win = s_red[0]; __syncthreads();
    if (win != ~0ULL && (unsigned)(win & 0xFFull) == (unsigned)tid){
      #pragma unroll
      for (int i = 0; i < 10; ++i){ if (s_ilist[tid*10+i] == win){ s_ilist[tid*10+i] = ~0ULL; break; } }
    }
    if (tid == 0 && win != ~0ULL) isum += ordflt(~(unsigned)(win >> 32));
    __syncthreads();
  }
  // global top-10 smallest costs
  for (int r = 0; r < 10; ++r){
    u64 m = ~0ULL;
    #pragma unroll
    for (int i = 0; i < 10; ++i){ u64 v = s_list[tid*10+i]; if (v < m) m = v; }
    s_red[tid] = m; __syncthreads();
    for (int off = 128; off > 0; off >>= 1){
      if (tid < off){ u64 o = s_red[tid + off]; if (o < s_red[tid]) s_red[tid] = o; }
      __syncthreads();
    }
    u64 win = s_red[0]; __syncthreads();
    if (win != ~0ULL && (unsigned)(win & 0xFFull) == (unsigned)tid){
      #pragma unroll
      for (int i = 0; i < 10; ++i){ if (s_list[tid*10+i] == win){ s_list[tid*10+i] = ~0ULL; break; } }
    }
    if (tid == 0) s_round[r] = win;
    __syncthreads();
  }
  if (tid == 0){
    int dynk = (int)isum; if (dynk < 1) dynk = 1; if (dynk > 10) dynk = 10;
    for (int r = 0; r < dynk; ++r){
      u64 w = s_round[r];
      if (w == ~0ULL) continue;
      int a = (int)((w >> 8) & 0xFFFFull);
      unsigned oc = (unsigned)(w >> 32);
      // conflict resolution: per-anchor min over (cost, gt)
      atomicMin(&winner[(size_t)b * NUMA + a], ((u64)oc << 32) | (u64)(unsigned)g);
    }
  }
}

// ---------------- K4: per-anchor losses -> per-image partial sums ----------------
__global__ void k_loss(const float* cls0, const float* cls1, const float* cls2,
                       const float* obj0, const float* obj1, const float* obj2,
                       const float* lmk0, const float* lmk1, const float* lmk2,
                       const float* __restrict__ gt_boxes,
                       const float* __restrict__ gt_lmk,
                       const float* __restrict__ boxes,
                       const u64* __restrict__ winner,
                       float* __restrict__ sums)
{
  __shared__ float sh[256];
  int b = blockIdx.y;
  int a = blockIdx.x * blockDim.x + threadIdx.x;
  float li = 0.f, lo = 0.f, lc = 0.f, nf = 0.f, lm = 0.f, vmc = 0.f;
  if (a < NUMA){
    int idx = b * NUMA + a;
    u64 w = winner[idx];
    bool fg = (w != ~0ULL);
    int lvl, x, y, W; float s; adec(a, lvl, x, y, s, W);
    const float* op = (lvl == 0) ? obj0 : (lvl == 1 ? obj1 : obj2);
    size_t o1 = ((size_t)b * W + y) * W + x;
    float ob = op[o1];
    lo = softplusf(ob) - ob * (fg ? 1.f : 0.f);
    if (fg){
      nf = 1.f;
      int g = (int)(w & 0xFFFFFFFFull);
      const float* gb = gt_boxes + ((size_t)b * NUMG + g) * 4;
      float gx0 = gb[0], gy0 = gb[1], gx1 = gb[2], gy1 = gb[3];
      const float* bp = boxes + (size_t)idx * 4;
      float b0 = bp[0], b1 = bp[1], b2 = bp[2], b3 = bp[3];
      float iw = fmaxf(fminf(gx1, b2) - fmaxf(gx0, b0), 0.f);
      float ih = fmaxf(fminf(gy1, b3) - fmaxf(gy0, b1), 0.f);
      float inter = iw * ih;
      float aa = (b2 - b0) * (b3 - b1), ab = (gx1 - gx0) * (gy1 - gy0);
      float uni = aa + ab - inter;
      float iou = inter / (uni + 1e-8f);
      float cw = fmaxf(fmaxf(gx1, b2) - fminf(gx0, b0), 0.f);
      float ch = fmaxf(fmaxf(gy1, b3) - fminf(gy0, b1), 0.f);
      float car = cw * ch;
      float giou = iou - (car - uni) / (car + 1e-8f);
      li = 1.f - giou;
      const float* cp = (lvl == 0) ? cls0 : (lvl == 1 ? cls1 : cls2);
      float cl = cp[o1];
      lc = softplusf(cl) - cl * iou;           // target = pred_iou
      const float* lp = (lvl == 0) ? lmk0 : (lvl == 1 ? lmk1 : lmk2);
      const float* gl = gt_lmk + ((size_t)b * NUMG + g) * 10;
      size_t plane = (size_t)W * W;
      #pragma unroll
      for (int k = 0; k < 5; ++k){
        float tx = gl[2*k], ty = gl[2*k+1];
        float vm = (tx >= 0.f && ty >= 0.f) ? 1.f : 0.f;
        float pxv = lp[((size_t)b * 10 + 2*k)     * plane + (size_t)y * W + x];
        float pyv = lp[((size_t)b * 10 + 2*k + 1) * plane + (size_t)y * W + x];
        float px = (pxv + x) * s, py = (pyv + y) * s;
        lm  += vm * (sl1(px - tx) + sl1(py - ty));
        vmc += vm;
      }
    }
  }
  float vals[6] = { li, lo, lc, nf, lm, vmc };
  for (int q = 0; q < 6; ++q){
    sh[threadIdx.x] = vals[q];
    __syncthreads();
    for (int off = 128; off > 0; off >>= 1){
      if (threadIdx.x < off) sh[threadIdx.x] += sh[threadIdx.x + off];
      __syncthreads();
    }
    if (threadIdx.x == 0) atomicAdd(&sums[b * 8 + q], sh[0]);
    __syncthreads();
  }
}

// ---------------- K5: WMMA reduction over the 16 images + finalize ----------------
__device__ __forceinline__ float loadq(const float* sums, int j, int img){
  float v0 = sums[img * 8 + (j & 7)];
  float v5 = sums[img * 8 + 5];
  float r = (j == 4) ? v0 / fmaxf(v5, 1.f) : v0;   // lmk normalized per image by vm count
  return (j < 5) ? r : 0.f;
}

__global__ void k_final(const float* __restrict__ sums, float* __restrict__ out)
{
  __shared__ float s_t[16];
  int lane = threadIdx.x;          // 0..31, one wave32
  int hi = lane >> 4, j = lane & 15;
  float tot = 0.f;
#if __has_builtin(__builtin_amdgcn_wmma_f32_16x16x4_f32)
  // D += ones(16x4) x B(4x16): column sums of 4-image slices; chain 4 to cover 16 images.
  v2f aop = { 1.f, 1.f };          // A all-ones (16x4 layout: lanes hold K pairs)
  v8f acc = {};
  for (int c4 = 0; c4 < 4; ++c4){
    v2f bop;
    bop[0] = loadq(sums, j, c4 * 4 + hi);       // B rows 0/1
    bop[1] = loadq(sums, j, c4 * 4 + 2 + hi);   // B rows 2/3
    acc = __builtin_amdgcn_wmma_f32_16x16x4_f32(false, aop, false, bop,
                                                (short)0, acc, false, false);
  }
  tot = acc[0];                    // lanes 0..15: D[0][j] = total of quantity j
#else
  // fallback: codegen-confirmed f16 WMMA ones-reduction
  v16h ah, bh;
  #pragma unroll
  for (int e = 0; e < 16; ++e) ah[e] = (_Float16)1.0f;
  #pragma unroll
  for (int e = 0; e < 16; ++e) bh[e] = (_Float16)loadq(sums, j, e);
  v8f acc = {};
  acc = __builtin_amdgcn_wmma_f32_16x16x32_f16(false, ah, false, bh,
                                               (short)0, acc, false, false);
  tot = acc[0];
#endif
  if (lane < 16) s_t[j] = tot;
  __syncthreads();
  if (lane == 0){
    float nfg  = fmaxf(s_t[3], 1.f);
    float liou = 5.f * s_t[0] / nfg;
    float lobj =       s_t[1] / nfg;
    float lcls =       s_t[2] / nfg;
    float llmk = 5.f * s_t[4] / nfg;
    out[0] = liou + lobj + lcls + llmk;
    out[1] = liou; out[2] = lobj; out[3] = lcls; out[4] = llmk;
  }
}

extern "C" void kernel_launch(void* const* d_in, const int* in_sizes, int n_in,
                              void* d_out, int out_size, void* d_ws, size_t ws_size,
                              hipStream_t stream)
{
  (void)in_sizes; (void)n_in; (void)out_size; (void)ws_size;
  const float* cls0 = (const float*)d_in[0];
  const float* reg0 = (const float*)d_in[1];
  const float* obj0 = (const float*)d_in[2];
  const float* lmk0 = (const float*)d_in[3];
  const float* cls1 = (const float*)d_in[4];
  const float* reg1 = (const float*)d_in[5];
  const float* obj1 = (const float*)d_in[6];
  const float* lmk1 = (const float*)d_in[7];
  const float* cls2 = (const float*)d_in[8];
  const float* reg2 = (const float*)d_in[9];
  const float* obj2 = (const float*)d_in[10];
  const float* lmk2 = (const float*)d_in[11];
  const float* gt_boxes = (const float*)d_in[12];
  const float* gt_lmk   = (const float*)d_in[13];
  const unsigned char* gt_valid = (const unsigned char*)d_in[15];
  float* out = (float*)d_out;

  char* ws = (char*)d_ws;
  size_t off = 0;
  float* boxes  = (float*)(ws + off); off += (size_t)NUMB * NUMA * 4 * sizeof(float);
  u64*   winner = (u64*)  (ws + off); off += (size_t)NUMB * NUMA * sizeof(u64);
  u64*   inboth = (u64*)  (ws + off); off += (size_t)NUMB * NUMA * sizeof(u64);
  unsigned* fgm = (unsigned*)(ws + off); off += (size_t)NUMB * NUMA * sizeof(unsigned);
  float* sums   = (float*)(ws + off); off += (size_t)NUMB * 8 * sizeof(float);

  dim3 blk(256);
  k_decode<<<dim3((NUMB * NUMA + 255) / 256), blk, 0, stream>>>(reg0, reg1, reg2, boxes, winner, sums);
  k_mask  <<<dim3((NUMA + 255) / 256, NUMB), blk, 0, stream>>>(gt_boxes, gt_valid, inboth, fgm);
  k_assign<<<dim3(NUMG, NUMB), blk, 0, stream>>>(cls0, cls1, cls2, obj0, obj1, obj2,
                                                 gt_boxes, gt_valid, boxes, fgm, inboth, winner);
  k_loss  <<<dim3((NUMA + 255) / 256, NUMB), blk, 0, stream>>>(cls0, cls1, cls2, obj0, obj1, obj2,
                                                 lmk0, lmk1, lmk2, gt_boxes, gt_lmk, boxes, winner, sums);
  k_final <<<1, 32, 0, stream>>>(sums, out);
}
